// SRNN_ALIF_40046275068554
// MI455X (gfx1250) — matrix operand order repaired and checked by reading, not compiled, not run on hardware
//
#include <hip/hip_runtime.h>
#include <math.h>

// ---------------- types ----------------
typedef __bf16 v8bf  __attribute__((ext_vector_type(8)));
typedef __bf16 v16bf __attribute__((ext_vector_type(16)));
typedef float  v8f   __attribute__((ext_vector_type(8)));

// ---------------- problem constants ----------------
#define NB   64
#define TT   100
#define INK  1024
#define HD   512
#define KCAT 1536
#define OUTS 11
#define MFF  (TT*NB)          // 6400 rows for feed-forward GEMMs
#define STN  (NB*HD)          // 32768 state elems per array

#define SHUF16(lo, hi) __builtin_shufflevector(lo, hi, \
    0,1,2,3,4,5,6,7,8,9,10,11,12,13,14,15)
#define WMMA_BF16(a, b, c) __builtin_amdgcn_wmma_f32_16x16x32_bf16( \
    false, a, false, b, (short)0, c, false, false)

// ======================================================================
// Software-pipelined 16(M)x64(N) WMMA tile MAC over K (K % 32 == 0).
// ap = A row (m0+lidx) + half*16     (row-major, ld = K)
// bp = W row (n0+lidx) + half*16    (row-major, ld = K; tile j at +j*16*K)
// Double-buffers the 2 A-loads + 4 B-loads so iteration k+1's loads
// overlap iteration k's 4 WMMAs (avoids s_wait_loadcnt 0 before each WMMA).
// ======================================================================
template <int K>
__device__ __forceinline__ void wmma_tile_16x64(const __bf16* __restrict__ ap,
                                                const __bf16* __restrict__ bp,
                                                v8f (&acc)[4]) {
    v8bf  alo = *(const v8bf*)(ap);
    v8bf  ahi = *(const v8bf*)(ap + 8);
    v16bf b0  = *(const v16bf*)(bp);
    v16bf b1  = *(const v16bf*)(bp + (size_t)16 * K);
    v16bf b2  = *(const v16bf*)(bp + (size_t)32 * K);
    v16bf b3  = *(const v16bf*)(bp + (size_t)48 * K);

    for (int k0 = 0; k0 < K - 32; k0 += 32) {
        v16bf a  = SHUF16(alo, ahi);
        v16bf c0 = b0, c1 = b1, c2 = b2, c3 = b3;
        const int kn = k0 + 32;
        // prefetch next K-step (in flight during the WMMAs below)
        alo = *(const v8bf*)(ap + kn);
        ahi = *(const v8bf*)(ap + kn + 8);
        b0  = *(const v16bf*)(bp + kn);
        b1  = *(const v16bf*)(bp + kn + (size_t)16 * K);
        b2  = *(const v16bf*)(bp + kn + (size_t)32 * K);
        b3  = *(const v16bf*)(bp + kn + (size_t)48 * K);
        acc[0] = WMMA_BF16(a, c0, acc[0]);
        acc[1] = WMMA_BF16(a, c1, acc[1]);
        acc[2] = WMMA_BF16(a, c2, acc[2]);
        acc[3] = WMMA_BF16(a, c3, acc[3]);
    }
    v16bf a = SHUF16(alo, ahi);
    acc[0] = WMMA_BF16(a, b0, acc[0]);
    acc[1] = WMMA_BF16(a, b1, acc[1]);
    acc[2] = WMMA_BF16(a, b2, acc[2]);
    acc[3] = WMMA_BF16(a, b3, acc[3]);
}

// ======================================================================
// 1) 4x4 max-pool, f32 -> bf16, layout xp[c][(t*64+n)*1024 + ph*32+pw]
// ======================================================================
__global__ void pool_kernel(const float* __restrict__ x,
                            __bf16* __restrict__ xp0,
                            __bf16* __restrict__ xp1) {
    int id = blockIdx.x * blockDim.x + threadIdx.x;      // 13,107,200 threads
    int pw = id & 31;  int r1 = id >> 5;
    int ph = r1 & 31;  int r2 = r1 >> 5;
    int c  = r2 & 1;   int r3 = r2 >> 1;                 // r3 in [0, 6400)
    int t  = r3 % TT;
    int n  = r3 / TT;
    const float* src = x + ((((size_t)(n * TT + t) * 2 + c) * 128 + ph * 4) * 128 + pw * 4);
    float mx = -3.402823466e38f;
#pragma unroll
    for (int r = 0; r < 4; ++r) {
        float4 v = *(const float4*)(src + r * 128);
        mx = fmaxf(mx, fmaxf(fmaxf(v.x, v.y), fmaxf(v.z, v.w)));
    }
    __bf16* dst = c ? xp1 : xp0;
    dst[(size_t)(t * NB + n) * INK + ph * 32 + pw] = (__bf16)mx;
}

// ======================================================================
// 2) weight prep: f32->bf16 for w1,w2; wcat = [w_i2h3 | w_h2h3] (512x1536);
//    bcat = b_i2h3 + b_h2h3
// ======================================================================
__global__ void prep_weights(const float* __restrict__ w1f, const float* __restrict__ w2f,
                             const float* __restrict__ w3f, const float* __restrict__ whf,
                             const float* __restrict__ b3f, const float* __restrict__ bhf,
                             __bf16* __restrict__ w1, __bf16* __restrict__ w2,
                             __bf16* __restrict__ wcat, float* __restrict__ bcat) {
    int id = blockIdx.x * blockDim.x + threadIdx.x;      // 512*1536 threads
    int k = id % KCAT;
    int h = id / KCAT;
    if (k < INK) {
        wcat[(size_t)h * KCAT + k] = (__bf16)w3f[(size_t)h * INK + k];
        w1[(size_t)h * INK + k]    = (__bf16)w1f[(size_t)h * INK + k];
        w2[(size_t)h * INK + k]    = (__bf16)w2f[(size_t)h * INK + k];
    } else {
        wcat[(size_t)h * KCAT + k] = (__bf16)whf[(size_t)h * HD + (k - INK)];
    }
    if (k == 0) bcat[h] = b3f[h] + bhf[h];
}

// ======================================================================
// 3) state init: st = [m1,s1,b1, m2,s2,b2, m3,s3,b3] each 64*512 f32
// ======================================================================
__global__ void init_state(float* __restrict__ st) {
    int i = blockIdx.x * blockDim.x + threadIdx.x;       // 32768 threads
    st[0 * STN + i] = 0.f; st[1 * STN + i] = 0.f; st[2 * STN + i] = 0.1f;
    st[3 * STN + i] = 0.f; st[4 * STN + i] = 0.f; st[5 * STN + i] = 0.1f;
    st[6 * STN + i] = 0.f; st[7 * STN + i] = 0.f; st[8 * STN + i] = 0.1f;
}

// ======================================================================
// 4) feed-forward GEMMs: out[m][h] = A[m][:] . W[h][:] + bias[h]
//    M=6400, N=512, K=1024.  One wave per 16x64 tile.
//    grid = (N/64, M/16, channel)
// ======================================================================
__global__ void gemm_ff(const __bf16* __restrict__ xp0, const __bf16* __restrict__ xp1,
                        const __bf16* __restrict__ w1,  const __bf16* __restrict__ w2,
                        const float* __restrict__ b1,   const float* __restrict__ b2,
                        float* __restrict__ in1,        float* __restrict__ in2) {
    const int c = blockIdx.z;
    const __bf16* A    = c ? xp1 : xp0;
    const __bf16* W    = c ? w2  : w1;
    const float*  bias = c ? b2  : b1;
    float*        out  = c ? in2 : in1;

    const int m0   = blockIdx.y * 16;
    const int n0   = blockIdx.x * 64;
    const int lane = threadIdx.x;
    const int half = lane >> 4;     // contiguous K-half (k = 16*half + s) for A and B alike
    const int lidx = lane & 15;     // A row / B,D column within tile

    v8f acc[4] = {v8f{}, v8f{}, v8f{}, v8f{}};
    const __bf16* ap = A + (size_t)(m0 + lidx) * INK + half * 16;
    const __bf16* bp = W + (size_t)(n0 + lidx) * INK + half * 16;
    wmma_tile_16x64<INK>(ap, bp, acc);

#pragma unroll
    for (int j = 0; j < 4; ++j) {
        const int n = n0 + j * 16 + lidx;
        const float bv = bias[n];
#pragma unroll
        for (int r = 0; r < 8; ++r)    // D: elem r -> row half*8+r, col lidx
            out[(size_t)(m0 + half * 8 + r) * HD + n] = acc[j][r] + bv;
    }
}

// ======================================================================
// 5) per-step ALIF for layers 1 & 2 (+ bf16 snapshot of old s3) into
//    sall[n][0..511]=s1, [512..1023]=s2, [1024..1535]=s3_old
// ======================================================================
__global__ void alif12_snapshot(const float* __restrict__ in1, const float* __restrict__ in2,
                                const float* __restrict__ ta1, const float* __restrict__ tm1,
                                const float* __restrict__ ta2, const float* __restrict__ tm2,
                                float* __restrict__ st, __bf16* __restrict__ sall, int t) {
    int id = blockIdx.x * blockDim.x + threadIdx.x;      // 3*32768 threads
    int seg = id >> 15;                                  // 0:L1  1:L2  2:s3 snap
    int r   = id & (STN - 1);
    int n   = r >> 9;
    int j   = r & (HD - 1);
    if (seg == 2) {
        sall[(size_t)n * KCAT + INK + j] = (__bf16)st[7 * STN + r];
        return;
    }
    const float* inb = seg ? in2 : in1;
    const float* ta  = seg ? ta2 : ta1;
    const float* tm  = seg ? tm2 : tm1;
    float* m = st + (seg ? 3 : 0) * STN;
    float* s = st + (seg ? 4 : 1) * STN;
    float* b = st + (seg ? 5 : 2) * STN;

    float inp   = inb[(size_t)(t * NB + n) * HD + j];
    float ro    = __expf(-1.0f / ta[j]);
    float alpha = __expf(-1.0f / tm[j]);
    float sold  = s[r];
    float bb    = ro * b[r] + (1.0f - ro) * sold;
    float Bth   = 0.1f + 1.8f * bb;
    float mnew  = m[r] * alpha + (1.0f - alpha) * inp - Bth * sold;
    float snew  = (mnew - Bth) > 0.0f ? 1.0f : 0.0f;
    m[r] = mnew; s[r] = snew; b[r] = bb;
    sall[(size_t)n * KCAT + seg * HD + j] = (__bf16)snew;
}

// ======================================================================
// 6) recurrent GEMM [64x1536]x[1536x512] with fused ALIF-3 epilogue.
//    grid = (512/64, 64/16), one wave / 16x64 tile, 48 K-steps.
// ======================================================================
__global__ void gemm_rec_alif3(const __bf16* __restrict__ sall, const __bf16* __restrict__ wcat,
                               const float* __restrict__ bcat,
                               const float* __restrict__ ta3, const float* __restrict__ tm3,
                               float* __restrict__ st) {
    float* m3 = st + 6 * STN;
    float* s3 = st + 7 * STN;
    float* b3 = st + 8 * STN;

    const int m0   = blockIdx.y * 16;
    const int n0   = blockIdx.x * 64;
    const int lane = threadIdx.x;
    const int half = lane >> 4;
    const int lidx = lane & 15;

    v8f acc[4] = {v8f{}, v8f{}, v8f{}, v8f{}};
    const __bf16* ap = sall + (size_t)(m0 + lidx) * KCAT + half * 16;
    const __bf16* bp = wcat + (size_t)(n0 + lidx) * KCAT + half * 16;
    wmma_tile_16x64<KCAT>(ap, bp, acc);

#pragma unroll
    for (int j = 0; j < 4; ++j) {
        const int n = n0 + j * 16 + lidx;
        const float ro    = __expf(-1.0f / ta3[n]);
        const float alpha = __expf(-1.0f / tm3[n]);
        const float bv    = bcat[n];
#pragma unroll
        for (int r = 0; r < 8; ++r) {
            const int m   = m0 + half * 8 + r;
            const int idx = m * HD + n;
            float inp  = acc[j][r] + bv;
            float sold = s3[idx];
            float bb   = ro * b3[idx] + (1.0f - ro) * sold;
            float Bth  = 0.1f + 1.8f * bb;
            float mnew = m3[idx] * alpha + (1.0f - alpha) * inp - Bth * sold;
            float snew = (mnew - Bth) > 0.0f ? 1.0f : 0.0f;
            m3[idx] = mnew; s3[idx] = snew; b3[idx] = bb;
        }
    }
}

// ======================================================================
// 7) head: out[n][o] = s3_final[n] . w_h2o3[o] + b_h2o3[o]   (f32 scalar)
// ======================================================================
__global__ void head_kernel(const float* __restrict__ st, const float* __restrict__ w,
                            const float* __restrict__ b, float* __restrict__ out) {
    int id = blockIdx.x * blockDim.x + threadIdx.x;
    if (id >= NB * OUTS) return;
    int o = id % OUTS;
    int n = id / OUTS;
    const float* s3 = st + 7 * STN + (size_t)n * HD;
    const float* wr = w + (size_t)o * HD;
    float acc = b[o];
    for (int j = 0; j < HD; ++j) acc += s3[j] * wr[j];
    out[id] = acc;
}

// ======================================================================
extern "C" void kernel_launch(void* const* d_in, const int* in_sizes, int n_in,
                              void* d_out, int out_size, void* d_ws, size_t ws_size,
                              hipStream_t stream) {
    (void)in_sizes; (void)n_in; (void)out_size; (void)ws_size;
    const float* x       = (const float*)d_in[0];
    const float* w_i2h1  = (const float*)d_in[1];
    const float* b_i2h1  = (const float*)d_in[2];
    const float* w_i2h2  = (const float*)d_in[3];
    const float* b_i2h2  = (const float*)d_in[4];
    const float* w_i2h3  = (const float*)d_in[5];
    const float* b_i2h3  = (const float*)d_in[6];
    const float* w_h2h3  = (const float*)d_in[7];
    const float* b_h2h3  = (const float*)d_in[8];
    const float* w_h2o3  = (const float*)d_in[9];
    const float* b_h2o3  = (const float*)d_in[10];
    const float* ta1     = (const float*)d_in[11];
    const float* ta2     = (const float*)d_in[12];
    const float* ta3     = (const float*)d_in[13];
    const float* tm1     = (const float*)d_in[14];
    const float* tm2     = (const float*)d_in[15];
    const float* tm3     = (const float*)d_in[16];

    char* ws = (char*)d_ws;
    size_t off = 0;
    auto alloc = [&](size_t bytes) -> void* {
        void* p = (void*)(ws + off);
        off += (bytes + 255) & ~(size_t)255;
        return p;
    };
    __bf16* xp0  = (__bf16*)alloc((size_t)MFF * INK * 2);        // 13.1 MB
    __bf16* xp1  = (__bf16*)alloc((size_t)MFF * INK * 2);        // 13.1 MB
    __bf16* w1   = (__bf16*)alloc((size_t)HD * INK * 2);         // 1 MB
    __bf16* w2   = (__bf16*)alloc((size_t)HD * INK * 2);         // 1 MB
    __bf16* wcat = (__bf16*)alloc((size_t)HD * KCAT * 2);        // 1.5 MB
    float*  bcat = (float*)alloc((size_t)HD * 4);
    float*  in1  = (float*)alloc((size_t)MFF * HD * 4);          // 13.1 MB
    float*  in2  = (float*)alloc((size_t)MFF * HD * 4);          // 13.1 MB
    __bf16* sall = (__bf16*)alloc((size_t)NB * KCAT * 2);        // 197 KB
    float*  st   = (float*)alloc((size_t)9 * STN * 4);           // 1.2 MB

    init_state<<<STN / 256, 256, 0, stream>>>(st);
    prep_weights<<<(HD * KCAT) / 256, 256, 0, stream>>>(
        w_i2h1, w_i2h2, w_i2h3, w_h2h3, b_i2h3, b_h2h3, w1, w2, wcat, bcat);
    pool_kernel<<<(NB * TT * 2 * 1024) / 256, 256, 0, stream>>>(x, xp0, xp1);
    gemm_ff<<<dim3(HD / 64, MFF / 16, 2), 32, 0, stream>>>(
        xp0, xp1, w1, w2, b_i2h1, b_i2h2, in1, in2);

    for (int t = 0; t < TT; ++t) {
        alif12_snapshot<<<(3 * STN) / 256, 256, 0, stream>>>(
            in1, in2, ta1, tm1, ta2, tm2, st, sall, t);
        gemm_rec_alif3<<<dim3(HD / 64, NB / 16), 32, 0, stream>>>(
            sall, wcat, bcat, ta3, tm3, st);
    }

    head_kernel<<<(NB * OUTS + 255) / 256, 256, 0, stream>>>(
        st, w_h2o3, b_h2o3, (float*)d_out);
}
